// Net_36077725286545
// MI455X (gfx1250) — compile-verified, matrix operand order checked by reading, not run on hardware
//
#include <hip/hip_runtime.h>
#include <hip/hip_bf16.h>
#include <stddef.h>

// Problem constants (from reference)
#define NET 500
#define DIM 10
#define NB  32
#define TT  8

// Workspace layout (bytes). Total ~133.4 MB required.
#define WS_MM    0                         // 32 uints: [0..7] ain_min, [8..15] ain_max, [16..23] anet_min, [24..31] anet_max
#define WS_ASWZ  1024                      // 8*2*16*32*16 bf16 = 262144 B  (pre-swizzled A fragments)
#define WS_AIN   (WS_ASWZ + 262144)        // T*N*5000 fp32 = 5,120,000 B  (raw a_in)
#define WS_ANET  (WS_AIN + 5120000)        // T*N*250000 bf16 = 128,000,000 B (raw relu a_net)

typedef __attribute__((ext_vector_type(16))) __bf16 v16bf;
typedef __attribute__((ext_vector_type(8)))  float  v8f;

__device__ __forceinline__ unsigned short f2bf(float f) {
    unsigned u = __float_as_uint(f);
    u += 0x7FFFu + ((u >> 16) & 1u);          // round-to-nearest-even
    return (unsigned short)(u >> 16);
}
__device__ __forceinline__ float bf2f(unsigned short h) {
    return __uint_as_float(((unsigned)h) << 16);
}
// sigmoid(100 * (norm01(raw) - 0.5)) with norm = (raw - mn) * inv
__device__ __forceinline__ float sig100(float raw, float mn, float inv) {
    float z = (raw - mn) * inv;
    return 1.f / (1.f + __expf(50.f - 100.f * z));
}

// 16-bit operand fragment swizzle (ISA §7.12.2, A 16x32):
// lane group 0 (lanes 0..15): holds kk in {0..7, 16..23}; group 1 (16..31): {8..15, 24..31}
// slot within lane: (kk&7) + 8*((kk>>4)&1)
__device__ __forceinline__ int frag_grp (int kk) { return (kk >> 3) & 1; }
__device__ __forceinline__ int frag_slot(int kk) { return (kk & 7) + (((kk >> 4) & 1) << 3); }

// ---------------------------------------------------------------------------
__global__ void k_init(unsigned* mm) {
    int tid = threadIdx.x;
    if (tid < 8) {
        mm[tid]      = 0x7F800000u;  // +inf  (ain min)
        mm[8 + tid]  = 0u;           // 0     (ain max; relu >= 0)
        mm[16 + tid] = 0x7F800000u;  // anet min
        mm[24 + tid] = 0u;           // anet max
    }
}

// ---------------------------------------------------------------------------
// Front end: per (t,n) block. MLP chains, raw a_in (+min/max), g chain,
// and writes g as pre-swizzled bf16 WMMA A-fragments.
__global__ void __launch_bounds__(256)
k_front(const float* __restrict__ x,
        const float* __restrict__ W1, const float* __restrict__ b1,
        const float* __restrict__ W2, const float* __restrict__ b2,
        const float* __restrict__ W3, const float* __restrict__ b3,
        const float* __restrict__ Wn1, const float* __restrict__ bn1,
        const float* __restrict__ Wn2, const float* __restrict__ bn2,
        unsigned* __restrict__ mm, float* __restrict__ ain_raw,
        unsigned short* __restrict__ aswz)
{
    int blk = blockIdx.x, tid = threadIdx.x;
    int t = blk >> 5, n = blk & 31;

    __shared__ float u[DIM];
    __shared__ float h1[300];
    __shared__ float h2[300];
    __shared__ float g1[NET];
    __shared__ float g2[512];
    __shared__ float rmn[256], rmx[256];

    if (tid < DIM) u[tid] = x[(n * DIM + tid) * TT + t];
    __syncthreads();

    if (tid < 300) {
        float a = b1[tid];
        for (int d = 0; d < DIM; ++d) a += W1[tid * DIM + d] * u[d];
        h1[tid] = fmaxf(a, 0.f);
    }
    __syncthreads();
    if (tid < 300) {
        float a = b2[tid];
        const float* wr = W2 + (size_t)tid * 300;
        for (int j = 0; j < 300; ++j) a += wr[j] * h1[j];
        h2[tid] = fmaxf(a, 0.f);
    }
    __syncthreads();

    // a_in raw: 5000 rows of dot-300, track relu min/max
    float lmn = 3.0e38f, lmx = 0.f;
    float* aout = ain_raw + (size_t)(t * NB + n) * (NET * DIM);
    for (int row = tid; row < NET * DIM; row += 256) {
        const float* wr = W3 + (size_t)row * 300;
        float a = b3[row];
        for (int j = 0; j < 300; ++j) a += wr[j] * h2[j];
        a = fmaxf(a, 0.f);
        aout[row] = a;
        lmn = fminf(lmn, a); lmx = fmaxf(lmx, a);
    }

    // g chain
    for (int i = tid; i < NET; i += 256) {
        float a = bn1[i];
        for (int d = 0; d < DIM; ++d) a += Wn1[i * DIM + d] * u[d];
        g1[i] = fmaxf(a, 0.f);
    }
    __syncthreads();
    for (int i = tid; i < 512; i += 256) {
        float a = 0.f;
        if (i < NET) {
            a = bn2[i];
            const float* wr = Wn2 + (size_t)i * NET;
            for (int j = 0; j < NET; ++j) a += wr[j] * g1[j];
            a = fmaxf(a, 0.f);
        }
        g2[i] = a;   // K padded 500 -> 512 with zeros
    }
    __syncthreads();

    // Emit A-fragment swizzle: (t, mtile=n/16, kstep, lane, slot)
    int mtile = n >> 4, rowm = n & 15;
    for (int k = tid; k < 512; k += 256) {
        int kstep = k >> 5, kk = k & 31;
        int lane = frag_grp(kk) ? (rowm + 16) : rowm;
        size_t dst = ((((size_t)(t * 2 + mtile) * 16 + kstep) * 32 + lane) << 4) + frag_slot(kk);
        aswz[dst] = f2bf(g2[k]);
    }

    // Block reduce a_in min/max -> per-t atomics (bit-ordered; relu >= 0)
    rmn[tid] = lmn; rmx[tid] = lmx;
    __syncthreads();
    for (int s = 128; s > 0; s >>= 1) {
        if (tid < s) {
            rmn[tid] = fminf(rmn[tid], rmn[tid + s]);
            rmx[tid] = fmaxf(rmx[tid], rmx[tid + s]);
        }
        __syncthreads();
    }
    if (tid == 0) {
        atomicMin(&mm[0 + t], __float_as_uint(rmn[0]));
        atomicMax(&mm[8 + t], __float_as_uint(rmx[0]));
    }
}

// ---------------------------------------------------------------------------
// Big GEMM: a_net[t][n][r] = relu(g_t[n] . Wn3[r] + bn3[r]), bf16 out.
// One block per 16-column Wn3 panel; Wn3 read from HBM exactly once.
// Wave w == timestep t; each wave does m-tiles {0,1} -> full 32-row output.
__global__ void __launch_bounds__(256)
k_gemm(const float* __restrict__ Wn3, const float* __restrict__ bn3,
       const unsigned short* __restrict__ aswz,
       unsigned* __restrict__ mm, unsigned short* __restrict__ anet)
{
    __shared__ __align__(32) unsigned short ldsB[16 * 32 * 16];  // 16KB B panel (bf16 fragments)
    int tid = threadIdx.x;
    size_t r0 = (size_t)blockIdx.x * 16;

    // Cooperative load: 16 Wn3 rows (= output cols) x 512 K (zero-padded), fp32->bf16,
    // stored in B-fragment swizzle.
    for (int idx = tid; idx < 16 * 512; idx += 256) {
        int c = idx >> 9;       // column within tile
        int k = idx & 511;
        float v = (k < NET) ? Wn3[(r0 + c) * NET + k] : 0.f;
        int kstep = k >> 5, kk = k & 31;
        int lane = frag_grp(kk) ? (c + 16) : c;
        ldsB[(((kstep << 5) + lane) << 4) + frag_slot(kk)] = f2bf(v);
    }
    __syncthreads();

    int t = tid >> 5, lane = tid & 31;
    const v16bf* A = (const v16bf*)aswz;
    v8f c0 = {}, c1 = {};
    for (int ks = 0; ks < 16; ++ks) {
        v16bf b  = *(const v16bf*)&ldsB[((ks << 5) + lane) << 4];
        v16bf a0 = A[((size_t)(t * 2 + 0) * 16 + ks) * 32 + lane];
        v16bf a1 = A[((size_t)(t * 2 + 1) * 16 + ks) * 32 + lane];
        c0 = __builtin_amdgcn_wmma_f32_16x16x32_bf16(false, a0, false, b, (short)0, c0, false, false);
        c1 = __builtin_amdgcn_wmma_f32_16x16x32_bf16(false, a1, false, b, (short)0, c1, false, false);
    }

    // Epilogue: +bn3, relu, store bf16, per-t min/max.
    int col = lane & 15, rsel = lane >> 4;
    size_t r = r0 + col;
    float bias = bn3[r];
    float lmn = 3.0e38f, lmx = 0.f;
    unsigned short* ob = anet + (size_t)t * NB * (NET * NET) + r;
    #pragma unroll
    for (int v = 0; v < 8; ++v) {
        int m = rsel ? (v + 8) : v;                 // C layout: lanes 0-15 -> M=v, 16-31 -> M=v+8
        float x0 = fmaxf(c0[v] + bias, 0.f);        // m-tile 0 -> n = m
        float x1 = fmaxf(c1[v] + bias, 0.f);        // m-tile 1 -> n = m+16
        ob[(size_t)m * (NET * NET)]        = f2bf(x0);
        ob[(size_t)(m + 16) * (NET * NET)] = f2bf(x1);
        lmn = fminf(lmn, fminf(x0, x1));
        lmx = fmaxf(lmx, fmaxf(x0, x1));
    }
    for (int o = 16; o; o >>= 1) {
        lmn = fminf(lmn, __shfl_xor(lmn, o, 32));
        lmx = fmaxf(lmx, __shfl_xor(lmx, o, 32));
    }
    if (lane == 0) {
        atomicMin(&mm[16 + t], __float_as_uint(lmn));
        atomicMax(&mm[24 + t], __float_as_uint(lmx));
    }
}

// ---------------------------------------------------------------------------
// Back end: per (t,n) block. Fused norm01+sigmoid, x_bias, two gated matvecs,
// and Wo projection.
__global__ void __launch_bounds__(256)
k_back(const float* __restrict__ x, const float* __restrict__ state0,
       const float* __restrict__ W_in, const float* __restrict__ W_net,
       const float* __restrict__ Wo, const float* __restrict__ bo,
       const unsigned* __restrict__ mm, const float* __restrict__ ain_raw,
       const unsigned short* __restrict__ anet, float* __restrict__ out)
{
    int blk = blockIdx.x, tid = threadIdx.x;
    int t = blk >> 5, n = blk & 31;

    __shared__ float st[NET], s1v[NET], s2v[NET], uu[DIM];

    float imn = __uint_as_float(mm[0 + t]);
    float imx = __uint_as_float(mm[8 + t]);
    float amn = __uint_as_float(mm[16 + t]);
    float amx = __uint_as_float(mm[24 + t]);
    float iinv = 1.f / (imx - imn);
    float ainv = 1.f / (amx - amn);

    if (tid < DIM) uu[tid] = x[(n * DIM + tid) * TT + t];
    for (int i = tid; i < NET; i += 256) st[i] = state0[(size_t)n * NET + i];
    __syncthreads();

    const unsigned short* abase = anet + (size_t)(t * NB + n) * (NET * NET);
    const float* ibase = ain_raw + (size_t)(t * NB + n) * (NET * DIM);

    float xb[2];
    #pragma unroll
    for (int p = 0; p < 2; ++p) {
        int i = tid + (p << 8);
        xb[p] = 0.f;
        if (i < NET) {
            float b = 0.f;
            for (int d = 0; d < DIM; ++d)
                b += sig100(ibase[i * DIM + d], imn, iinv) * W_in[i * DIM + d] * uu[d];
            xb[p] = b;
            const unsigned short* ar = abase + (size_t)i * NET;
            const float* wr = W_net + (size_t)i * NET;
            float acc = 0.f;
            for (int j = 0; j < NET; ++j)
                acc += sig100(bf2f(ar[j]), amn, ainv) * wr[j] * st[j];
            s1v[i] = fmaxf(acc + b, 0.f);
        }
    }
    __syncthreads();
    #pragma unroll
    for (int p = 0; p < 2; ++p) {
        int i = tid + (p << 8);
        if (i < NET) {
            const unsigned short* ar = abase + (size_t)i * NET;
            const float* wr = W_net + (size_t)i * NET;
            float acc = 0.f;
            for (int j = 0; j < NET; ++j)
                acc += sig100(bf2f(ar[j]), amn, ainv) * wr[j] * s1v[j];
            s2v[i] = fmaxf(acc + xb[p], 0.f);
        }
    }
    __syncthreads();
    if (tid < DIM) {
        float o = bo[tid];
        const float* wr = Wo + (size_t)tid * NET;
        for (int i = 0; i < NET; ++i) o += wr[i] * s2v[i];
        out[(n * DIM + tid) * TT + t] = o;
    }
}

// ---------------------------------------------------------------------------
extern "C" void kernel_launch(void* const* d_in, const int* in_sizes, int n_in,
                              void* d_out, int out_size, void* d_ws, size_t ws_size,
                              hipStream_t stream) {
    (void)in_sizes; (void)n_in; (void)out_size; (void)ws_size;
    const float* x      = (const float*)d_in[0];
    const float* state0 = (const float*)d_in[1];
    const float* W1  = (const float*)d_in[2];
    const float* b1  = (const float*)d_in[3];
    const float* W2  = (const float*)d_in[4];
    const float* b2  = (const float*)d_in[5];
    const float* W3  = (const float*)d_in[6];
    const float* b3  = (const float*)d_in[7];
    const float* Wn1 = (const float*)d_in[8];
    const float* bn1 = (const float*)d_in[9];
    const float* Wn2 = (const float*)d_in[10];
    const float* bn2 = (const float*)d_in[11];
    const float* Wn3 = (const float*)d_in[12];
    const float* bn3 = (const float*)d_in[13];
    const float* W_in  = (const float*)d_in[14];
    const float* W_net = (const float*)d_in[15];
    const float* Wo  = (const float*)d_in[16];
    const float* bo  = (const float*)d_in[17];
    float* out = (float*)d_out;

    char* ws = (char*)d_ws;
    unsigned*       mm   = (unsigned*)(ws + WS_MM);
    unsigned short* aswz = (unsigned short*)(ws + WS_ASWZ);
    float*          ain  = (float*)(ws + WS_AIN);
    unsigned short* anet = (unsigned short*)(ws + WS_ANET);

    k_init <<<1, 32, 0, stream>>>(mm);
    k_front<<<TT * NB, 256, 0, stream>>>(x, W1, b1, W2, b2, W3, b3,
                                         Wn1, bn1, Wn2, bn2, mm, ain, aswz);
    k_gemm <<<(NET * NET) / 16, 256, 0, stream>>>(Wn3, bn3, aswz, mm, anet);
    k_back <<<TT * NB, 256, 0, stream>>>(x, state0, W_in, W_net, Wo, bo,
                                         mm, ain, anet, out);
}
